// Encoder_73675868996408
// MI455X (gfx1250) — compile-verified, matrix operand order checked by reading, not run on hardware
//
#include <hip/hip_runtime.h>
#include <hip/hip_bf16.h>

typedef __attribute__((ext_vector_type(16))) _Float16 v16h;
typedef __attribute__((ext_vector_type(8)))  _Float16 v8h;
typedef __attribute__((ext_vector_type(8)))  float    v8f;

#define WMMA_F16(a, b, c) \
  __builtin_amdgcn_wmma_f32_16x16x32_f16(false, (a), false, (b), (short)0, (c), false, false)

// -------------------------------------------------------------------------
// Load a 16x32 (MxK) f16 fragment from a row-major matrix (leading dim `ld`
// in halves) following the CDNA5 16-bit A-matrix VGPR layout:
//   lanes 0-15  : M=0..15, K = 0..7   (regs 0-3) and K = 16..23 (regs 4-7)
//   lanes 16-31 : M=0..15, K = 8..15  (regs 0-3) and K = 24..31 (regs 4-7)
// B operands are supplied as B^T with the identical layout (N plays M).
// -------------------------------------------------------------------------
__device__ __forceinline__ v16h load_frag(const _Float16* base, int ld,
                                          int row0, int k0, int lane) {
  int r = lane & 15;
  int g = (lane >> 4) << 3;
  const _Float16* p = base + (row0 + r) * ld + k0 + g;
  v8h lo = *(const v8h*)(p);
  v8h hi = *(const v8h*)(p + 16);
  v16h out;
#pragma unroll
  for (int i = 0; i < 8; ++i) { out[i] = lo[i]; out[8 + i] = hi[i]; }
  return out;
}

__device__ __forceinline__ float sigm(float v) { return 1.f / (1.f + __expf(-v)); }

// -------------------------------------------------------------------------
// Row LayerNorm over 256 cols for 32 rows held in LDS (biased variance),
// writes normalized f32 back and an f16 copy for the next WMMA A-operand.
// -------------------------------------------------------------------------
__device__ __forceinline__ void ln_rows(float* sOup, _Float16* sOupH,
                                        float* sRed, float* sStat, int tid) {
  int l = tid >> 2, seg = tid & 3;
  float s1 = 0.f, s2 = 0.f;
  for (int c = seg * 64; c < seg * 64 + 64; ++c) {
    float v = sOup[l * 256 + c]; s1 += v; s2 += v * v;
  }
  sRed[(l * 4 + seg) * 2] = s1; sRed[(l * 4 + seg) * 2 + 1] = s2;
  __syncthreads();
  if (tid < 32) {
    float S1 = 0.f, S2 = 0.f;
    for (int j = 0; j < 4; ++j) { S1 += sRed[(tid * 4 + j) * 2]; S2 += sRed[(tid * 4 + j) * 2 + 1]; }
    float mean = S1 * (1.f / 256.f);
    float var  = S2 * (1.f / 256.f) - mean * mean;
    sStat[tid * 2]     = mean;
    sStat[tid * 2 + 1] = rsqrtf(fmaxf(var, 0.f) + 1e-5f);
  }
  __syncthreads();
  for (int c = seg * 64; c < seg * 64 + 64; ++c) {
    float v = (sOup[l * 256 + c] - sStat[l * 2]) * sStat[l * 2 + 1];
    sOup[l * 256 + c]  = v;
    sOupH[l * 256 + c] = (_Float16)v;
  }
  __syncthreads();
}

// -------------------------------------------------------------------------
// Weight prep: f32 -> f16 + transpose to [N][K] so WMMA B-fragments load
// contiguously.  Also packs the 4 quan matrices into one [120][256] f32.
// -------------------------------------------------------------------------
__global__ void encoder_prep(const float* __restrict__ fc1_w,
                             const float* __restrict__ att_w,
                             const float* __restrict__ att_ow,
                             const float* __restrict__ qw0, const float* __restrict__ qw1,
                             const float* __restrict__ qw2, const float* __restrict__ qw3,
                             const float* __restrict__ qb0, const float* __restrict__ qb1,
                             const float* __restrict__ qb2, const float* __restrict__ qb3,
                             _Float16* __restrict__ fc1_wT,
                             _Float16* __restrict__ att_wT,
                             _Float16* __restrict__ att_owT,
                             float* __restrict__ qwT, float* __restrict__ qbc) {
  int idx = blockIdx.x * 256 + threadIdx.x;
  if (idx < 98304) {                       // fc1_wT[256][384] <- fc1_w[384][256]
    int n = idx / 384, k = idx % 384;
    fc1_wT[idx] = (_Float16)fc1_w[k * 256 + n];
  } else if (idx < 245760) {               // att_wT[3][192][256] <- att_w[3][256][192]
    int t = idx - 98304;
    int i = t / 49152, rem = t % 49152;
    int n = rem / 256, k = rem % 256;
    att_wT[t] = (_Float16)att_w[i * 49152 + k * 192 + n];
  } else if (idx < 294912) {               // att_owT[3][256][64] <- att_ow[3][64][256]
    int t = idx - 245760;
    int i = t / 16384, rem = t % 16384;
    int n = rem / 64, k = rem % 64;
    att_owT[t] = (_Float16)att_ow[i * 16384 + k * 256 + n];
  } else if (idx < 325632) {               // qwT[120][256] (f32, packed)
    int t = idx - 294912;
    int j = t / 256, k = t % 256;
    int i, c;
    if (j < 35)      { i = 0; c = j; }
    else if (j < 67) { i = 1; c = j - 35; }
    else if (j < 95) { i = 2; c = j - 67; }
    else             { i = 3; c = j - 95; }
    const float* qw = (i == 0) ? qw0 : (i == 1) ? qw1 : (i == 2) ? qw2 : qw3;
    const int d     = (i == 0) ? 35  : (i == 1) ? 32  : (i == 2) ? 28  : 25;
    qwT[t] = qw[k * d + c];
  } else if (idx < 325752) {               // qbc[120]
    int j = idx - 325632;
    int i, c;
    if (j < 35)      { i = 0; c = j; }
    else if (j < 67) { i = 1; c = j - 35; }
    else if (j < 95) { i = 2; c = j - 67; }
    else             { i = 3; c = j - 95; }
    const float* qb = (i == 0) ? qb0 : (i == 1) ? qb1 : (i == 2) ? qb2 : qb3;
    qbc[j] = qb[c];
  }
}

// -------------------------------------------------------------------------
// Main fused encoder: one workgroup (4 wave32s) per batch element.
// All activations live in LDS; all GEMMs are v_wmma_f32_16x16x32_f16.
// Writes raw (pre-batchnorm) q to out[b][0..119] and indices to [120..123].
// -------------------------------------------------------------------------
__global__ __launch_bounds__(128) void encoder_main(
    const float* __restrict__ x, const float* __restrict__ PE,
    const float* __restrict__ fc1_b, const float* __restrict__ att_b,
    const float* __restrict__ att_ob,
    const _Float16* __restrict__ fc1_wT, const _Float16* __restrict__ att_wT,
    const _Float16* __restrict__ att_owT,
    const float* __restrict__ qwT, const float* __restrict__ qbc,
    float* __restrict__ out) {
  __shared__ __align__(16) _Float16 sA[32 * 384];     // fc1 A-matrix (f16)
  __shared__ __align__(16) float    sOup[32 * 256];   // running activations (f32)
  __shared__ __align__(16) _Float16 sOupH[32 * 256];  // f16 copy for WMMA A
  __shared__ __align__(16) _Float16 sQKV[32 * 192];
  __shared__ __align__(16) _Float16 sVT[2 * 32 * 32]; // V^T per head: [h][d][m]
  __shared__ __align__(16) float    sS[2 * 32 * 32];  // attention scores
  __shared__ __align__(16) _Float16 sAtt[2 * 32 * 32];
  __shared__ __align__(16) _Float16 sNV[32 * 64];
  __shared__ float sRed[256];
  __shared__ float sStat[64];
  __shared__ float sStd[32];
  __shared__ int   sIdx[4];

  const int tid  = threadIdx.x;
  const int lane = tid & 31;
  const int wave = tid >> 5;
  const int bi   = blockIdx.x;

  // ---------------- Phase 0: load x, build [32][384] input, std + top-4 ----
  const float* x0 = x + (size_t)bi * 2 * 32 * 128;
  const float* x1 = x0 + 32 * 128;
  {
    int l = tid >> 2, seg = tid & 3;
    float s1 = 0.f, s2 = 0.f;
    for (int f = seg * 32; f < seg * 32 + 32; ++f) {
      float a = x0[l * 128 + f];
      float c = x1[l * 128 + f];
      float m = sqrtf(a * a + c * c);
      sA[l * 384 + f]       = (_Float16)a;
      sA[l * 384 + 128 + f] = (_Float16)c;
      sA[l * 384 + 256 + f] = (_Float16)m;
      s1 += m; s2 += m * m;
    }
    sRed[(l * 4 + seg) * 2] = s1; sRed[(l * 4 + seg) * 2 + 1] = s2;
  }
  __syncthreads();
  if (tid < 32) {
    float S1 = 0.f, S2 = 0.f;
    for (int j = 0; j < 4; ++j) { S1 += sRed[(tid * 4 + j) * 2]; S2 += sRed[(tid * 4 + j) * 2 + 1]; }
    float mean = S1 * (1.f / 128.f);
    float var  = (S2 - 128.f * mean * mean) * (1.f / 127.f);   // unbiased (ddof=1)
    sStd[tid] = sqrtf(fmaxf(var, 0.f));
  }
  __syncthreads();
  if (tid == 0) {                                 // stable top-4 (ties -> lower index)
    unsigned taken = 0;
    for (int j = 0; j < 4; ++j) {
      float best = -3.0e38f; int bsel = 0;
      for (int m = 0; m < 32; ++m)
        if (!((taken >> m) & 1u) && sStd[m] > best) { best = sStd[m]; bsel = m; }
      sIdx[j] = bsel; taken |= 1u << bsel;
    }
  }
  __syncthreads();

  // ---------------- Phase 1: fc1 [32,384]x[384,256] + bias + PE + LN -------
  {
    v8f acc[2][4] = {};
    for (int k0 = 0; k0 < 384; k0 += 32) {
      v16h a0 = load_frag(sA, 384, 0, k0, lane);
      v16h a1 = load_frag(sA, 384, 16, k0, lane);
#pragma unroll
      for (int j = 0; j < 4; ++j) {
        v16h bw = load_frag(fc1_wT, 384, (wave * 4 + j) * 16, k0, lane);
        acc[0][j] = WMMA_F16(a0, bw, acc[0][j]);
        acc[1][j] = WMMA_F16(a1, bw, acc[1][j]);
      }
    }
    int n = lane & 15, mb = (lane >> 4) << 3;
#pragma unroll
    for (int j = 0; j < 4; ++j) {
      int ncol = (wave * 4 + j) * 16 + n;
      float bias = fc1_b[ncol];
#pragma unroll
      for (int mt = 0; mt < 2; ++mt)
#pragma unroll
        for (int r = 0; r < 8; ++r) {
          int row = mt * 16 + mb + r;
          sOup[row * 256 + ncol] = acc[mt][j][r] + bias + PE[row * 256 + ncol];
        }
    }
  }
  __syncthreads();
  ln_rows(sOup, sOupH, sRed, sStat, tid);

  // ---------------- Phase 2: three attention layers ------------------------
  for (int layer = 0; layer < 3; ++layer) {
    // (a) qkv = swish(oup @ w + b)
    {
      const _Float16* wT = att_wT + layer * 192 * 256;
      v8f acc[2][3] = {};
      for (int k0 = 0; k0 < 256; k0 += 32) {
        v16h a0 = load_frag(sOupH, 256, 0, k0, lane);
        v16h a1 = load_frag(sOupH, 256, 16, k0, lane);
#pragma unroll
        for (int j = 0; j < 3; ++j) {
          v16h bw = load_frag(wT, 256, (wave * 3 + j) * 16, k0, lane);
          acc[0][j] = WMMA_F16(a0, bw, acc[0][j]);
          acc[1][j] = WMMA_F16(a1, bw, acc[1][j]);
        }
      }
      int n = lane & 15, mb = (lane >> 4) << 3;
#pragma unroll
      for (int j = 0; j < 3; ++j) {
        int ncol = (wave * 3 + j) * 16 + n;
        float bias = att_b[layer * 192 + ncol];
#pragma unroll
        for (int mt = 0; mt < 2; ++mt)
#pragma unroll
          for (int r = 0; r < 8; ++r) {
            int row = mt * 16 + mb + r;
            float v = acc[mt][j][r] + bias;
            v = v * sigm(v);
            _Float16 h = (_Float16)v;
            sQKV[row * 192 + ncol] = h;
            int hh = ncol / 96, w96 = ncol % 96;
            if (w96 >= 64) sVT[hh * 1024 + (w96 - 64) * 32 + row] = h;  // V^T
          }
      }
    }
    __syncthreads();
    // (b) S = Q K^T per head (8 WMMA jobs over 4 waves)
#pragma unroll
    for (int jj = 0; jj < 2; ++jj) {
      int job = wave * 2 + jj;
      int h = job >> 2, mt = (job >> 1) & 1, nt = job & 1;
      v16h a  = load_frag(sQKV + h * 96,      192, mt * 16, 0, lane);   // Q rows
      v16h bw = load_frag(sQKV + h * 96 + 32, 192, nt * 16, 0, lane);   // K rows = B^T
      v8f c = {};
      c = WMMA_F16(a, bw, c);
      int n = lane & 15, mb = (lane >> 4) << 3;
#pragma unroll
      for (int r = 0; r < 8; ++r)
        sS[h * 1024 + (mt * 16 + mb + r) * 32 + nt * 16 + n] = c[r];
    }
    __syncthreads();
    // softmax over m (one thread per (head,row))
    if (tid < 64) {
      const float* row = &sS[tid * 32];
      float mx = row[0];
      for (int m = 1; m < 32; ++m) mx = fmaxf(mx, row[m]);
      float sum = 0.f;
      for (int m = 0; m < 32; ++m) sum += __expf(row[m] - mx);
      float inv = 1.f / sum;
      for (int m = 0; m < 32; ++m)
        sAtt[tid * 32 + m] = (_Float16)(__expf(row[m] - mx) * inv);
    }
    __syncthreads();
    // (c) nv = att @ V per head
#pragma unroll
    for (int jj = 0; jj < 2; ++jj) {
      int job = wave * 2 + jj;
      int h = job >> 2, mt = (job >> 1) & 1, nt = job & 1;
      v16h a  = load_frag(sAtt + h * 1024, 32, mt * 16, 0, lane);
      v16h bw = load_frag(sVT  + h * 1024, 32, nt * 16, 0, lane);       // (V^T) = B^T
      v8f c = {};
      c = WMMA_F16(a, bw, c);
      int n = lane & 15, mb = (lane >> 4) << 3;
#pragma unroll
      for (int r = 0; r < 8; ++r) {
        int w = mt * 16 + mb + r, d = nt * 16 + n;
        sNV[w * 64 + h * 32 + d] = (_Float16)c[r];
      }
    }
    __syncthreads();
    // (d) out-proj + swish + residual + LN
    {
      const _Float16* owT = att_owT + layer * 256 * 64;
      v8f acc[2][4] = {};
#pragma unroll
      for (int k0 = 0; k0 < 64; k0 += 32) {
        v16h a0 = load_frag(sNV, 64, 0, k0, lane);
        v16h a1 = load_frag(sNV, 64, 16, k0, lane);
#pragma unroll
        for (int j = 0; j < 4; ++j) {
          v16h bw = load_frag(owT, 64, (wave * 4 + j) * 16, k0, lane);
          acc[0][j] = WMMA_F16(a0, bw, acc[0][j]);
          acc[1][j] = WMMA_F16(a1, bw, acc[1][j]);
        }
      }
      int n = lane & 15, mb = (lane >> 4) << 3;
#pragma unroll
      for (int j = 0; j < 4; ++j) {
        int ncol = (wave * 4 + j) * 16 + n;
        float bias = att_ob[layer * 256 + ncol];
#pragma unroll
        for (int mt = 0; mt < 2; ++mt)
#pragma unroll
          for (int r = 0; r < 8; ++r) {
            int row = mt * 16 + mb + r;
            float v = acc[mt][j][r] + bias;
            v = v * sigm(v);
            sOup[row * 256 + ncol] += v;            // residual add
          }
      }
    }
    __syncthreads();
    ln_rows(sOup, sOupH, sRed, sStat, tid);
  }

  // ---------------- Phase 3: gather top-4 rows, quan projections -----------
  if (tid < 120) {
    int i4 = (tid < 35) ? 0 : (tid < 67) ? 1 : (tid < 95) ? 2 : 3;
    int row = sIdx[i4];
    const float* wrow = qwT + tid * 256;
    float acc = qbc[tid];
    for (int k = 0; k < 256; ++k) acc += sOup[row * 256 + k] * wrow[k];
    out[(size_t)bi * 124 + tid] = acc;              // raw; batch-norm in pass 3
  }
  if (tid >= 120 && tid < 124) {
    out[(size_t)bi * 124 + tid] = (float)sIdx[tid - 120];
  }
}

// -------------------------------------------------------------------------
// Pass 3: per-column batch normalization + sigmoid over B=8192 (in place).
// -------------------------------------------------------------------------
__global__ __launch_bounds__(256) void encoder_colnorm(float* __restrict__ out) {
  __shared__ float red[512];
  int col = blockIdx.x;
  float s1 = 0.f, s2 = 0.f;
  for (int b = threadIdx.x; b < 8192; b += 256) {
    float v = out[(size_t)b * 124 + col]; s1 += v; s2 += v * v;
  }
  red[threadIdx.x] = s1; red[256 + threadIdx.x] = s2;
  __syncthreads();
  for (int off = 128; off > 0; off >>= 1) {
    if (threadIdx.x < off) {
      red[threadIdx.x]       += red[threadIdx.x + off];
      red[256 + threadIdx.x] += red[256 + threadIdx.x + off];
    }
    __syncthreads();
  }
  float mean = red[0] * (1.f / 8192.f);
  float var  = red[256] * (1.f / 8192.f) - mean * mean;
  float inv  = rsqrtf(fmaxf(var, 0.f) + 1e-5f);
  for (int b = threadIdx.x; b < 8192; b += 256) {
    float v = out[(size_t)b * 124 + col];
    out[(size_t)b * 124 + col] = 1.f / (1.f + __expf(-(v - mean) * inv));
  }
}

extern "C" void kernel_launch(void* const* d_in, const int* in_sizes, int n_in,
                              void* d_out, int out_size, void* d_ws, size_t ws_size,
                              hipStream_t stream) {
  (void)in_sizes; (void)n_in; (void)out_size; (void)ws_size;
  const float* x      = (const float*)d_in[0];
  const float* PE     = (const float*)d_in[1];
  const float* fc1_w  = (const float*)d_in[2];
  const float* fc1_b  = (const float*)d_in[3];
  const float* att_w  = (const float*)d_in[4];
  const float* att_b  = (const float*)d_in[5];
  const float* att_ow = (const float*)d_in[6];
  const float* att_ob = (const float*)d_in[7];
  const float* qw0 = (const float*)d_in[8];  const float* qb0 = (const float*)d_in[9];
  const float* qw1 = (const float*)d_in[10]; const float* qb1 = (const float*)d_in[11];
  const float* qw2 = (const float*)d_in[12]; const float* qb2 = (const float*)d_in[13];
  const float* qw3 = (const float*)d_in[14]; const float* qb3 = (const float*)d_in[15];

  char* ws = (char*)d_ws;
  _Float16* fc1_wT  = (_Float16*)(ws + 0);        // 256*384 f16 = 196608 B
  _Float16* att_wT  = (_Float16*)(ws + 196608);   // 3*192*256 f16 = 294912 B
  _Float16* att_owT = (_Float16*)(ws + 491520);   // 3*256*64 f16 = 98304 B
  float*    qwT     = (float*)(ws + 589824);      // 120*256 f32 = 122880 B
  float*    qbc     = (float*)(ws + 712704);      // 120 f32

  encoder_prep<<<1273, 256, 0, stream>>>(fc1_w, att_w, att_ow,
                                         qw0, qw1, qw2, qw3,
                                         qb0, qb1, qb2, qb3,
                                         fc1_wT, att_wT, att_owT, qwT, qbc);
  encoder_main<<<8192, 128, 0, stream>>>(x, PE, fc1_b, att_b, att_ob,
                                         fc1_wT, att_wT, att_owT, qwT, qbc,
                                         (float*)d_out);
  encoder_colnorm<<<120, 256, 0, stream>>>((float*)d_out);
}
